// SparseMoE_4947802325412
// MI455X (gfx1250) — matrix operand order repaired and checked by reading, not compiled, not run on hardware
//
#include <hip/hip_runtime.h>
#include <hip/hip_bf16.h>

// ---------------------------------------------------------------------------
// Sparse MoE (top-2 of 8 experts), fp32 in/out, for MI455X (gfx1250).
// Strategy: split-bf16 WMMA GEMM (hi/lo decomposition, 3x v_wmma_f32_16x16x32_bf16
// per K-chunk) with per-expert token routing, 64-row-aligned expert segments,
// and atomic f32 scatter-add into the output (exactly 2 contributions/element).
// ---------------------------------------------------------------------------

#define NT   16384           // tokens (BATCH*SEQ)
#define DM   1024            // d_model
#define DI   4096            // d_inner
#define NE   8               // experts
#define NROW (2*NT)          // 32768 routed rows (top-2)
#define RMAX (NROW + NE*64)  // 33280: segments padded to 64 rows

// ---- workspace layout (int-indexed header, then bf16 planes) --------------
// wsi[0..7]   counts        wsi[8..15]  offsets
// wsf[16..23] sums          wsf[24..31] sumsq
#define TOKE_OFF 1024
#define TOKR_OFF (TOKE_OFF + NROW)
#define ROWT_OFF (TOKR_OFF + NROW)
#define ROWE_OFF (ROWT_OFF + RMAX)
#define INT_END  (ROWE_OFF + RMAX)                    // 133120 ints = 532480 B
#define XH_BYTE  ((size_t)INT_END * 4)
#define XL_BYTE  (XH_BYTE + (size_t)NT * DM * 2)
#define WH_BYTE  (XL_BYTE + (size_t)NT * DM * 2)
#define WL_BYTE  (WH_BYTE + (size_t)NE * DI * DM * 2)
// total ws needed: WL_BYTE + NE*DI*DM*2 ~= 202 MB

typedef float          f32x4  __attribute__((ext_vector_type(4)));
typedef unsigned int   u32x4  __attribute__((ext_vector_type(4)));
typedef unsigned short u16x4  __attribute__((ext_vector_type(4)));
typedef __attribute__((ext_vector_type(16))) __bf16 v16bf;
typedef __attribute__((ext_vector_type(8)))  float  v8f;

union Frag { u32x4 u[2]; v16bf v; };

__device__ __forceinline__ unsigned short f2bf(float f) {
  unsigned u = __float_as_uint(f);
  u += 0x7FFFu + ((u >> 16) & 1u);      // round-to-nearest-even
  return (unsigned short)(u >> 16);
}
__device__ __forceinline__ float bf2f(unsigned short h) {
  return __uint_as_float(((unsigned)h) << 16);
}

// ---------------------------------------------------------------------------
// 0) init: zero out[] (atomic-add target), reset routing scratch, rows = -1
// ---------------------------------------------------------------------------
__global__ void __launch_bounds__(256) moe_init_kernel(float* __restrict__ out,
                                                       int* __restrict__ wsi) {
  long i = (long)blockIdx.x * blockDim.x + threadIdx.x;
  long stride = (long)gridDim.x * blockDim.x;
  f32x4 z = {0.f, 0.f, 0.f, 0.f};
  for (long j = i; j < (long)NT * DI / 4; j += stride) ((f32x4*)out)[j] = z;
  for (long j = i; j < RMAX; j += stride) {
    wsi[ROWT_OFF + j] = -1;
    wsi[ROWE_OFF + j] = -1;
  }
  if (i < 8) {
    wsi[i] = 0;
    ((float*)wsi)[16 + i] = 0.f;
    ((float*)wsi)[24 + i] = 0.f;
  }
}

// ---------------------------------------------------------------------------
// 1) split fp32 -> bf16 hi/lo planes (x and expert_w), done once per call
// ---------------------------------------------------------------------------
__global__ void __launch_bounds__(256) moe_split_kernel(const float* __restrict__ src,
                                                        unsigned short* __restrict__ hi,
                                                        unsigned short* __restrict__ lo,
                                                        long n4) {
  long i = (long)blockIdx.x * blockDim.x + threadIdx.x;
  long stride = (long)gridDim.x * blockDim.x;
  for (; i < n4; i += stride) {
    f32x4 v = ((const f32x4*)src)[i];
    u16x4 h, l;
#pragma unroll
    for (int c = 0; c < 4; ++c) {
      unsigned short hh = f2bf(v[c]);
      h[c] = hh;
      l[c] = f2bf(v[c] - bf2f(hh));
    }
    ((u16x4*)hi)[i] = h;
    ((u16x4*)lo)[i] = l;
  }
}

// ---------------------------------------------------------------------------
// 2) gating: 1 wave32 per token. dot -> softmax -> top-2 -> rank + variance sums
// ---------------------------------------------------------------------------
__global__ void __launch_bounds__(256) moe_gating_kernel(
    const float* __restrict__ x, const float* __restrict__ gw,
    const float* __restrict__ gb, int* __restrict__ wsi) {
  __shared__ float s_p[8][NE], s_p2[8][NE];
  const int tid = threadIdx.x;
  const int wave = tid >> 5, lane = tid & 31;
  const int t = blockIdx.x * 8 + wave;
  const float* xr = x + (size_t)t * DM;

  float acc[NE];
#pragma unroll
  for (int e = 0; e < NE; ++e) acc[e] = 0.f;
  for (int i = lane; i < DM; i += 32) {
    float xv = xr[i];
#pragma unroll
    for (int e = 0; e < NE; ++e) acc[e] += xv * gw[e * DM + i];
  }
#pragma unroll
  for (int e = 0; e < NE; ++e)
    for (int off = 16; off > 0; off >>= 1) acc[e] += __shfl_xor(acc[e], off, 32);

  if (lane == 0) {
    float p[NE];
    float mx = -1e30f;
#pragma unroll
    for (int e = 0; e < NE; ++e) { p[e] = acc[e] + gb[e]; mx = fmaxf(mx, p[e]); }
    float s = 0.f;
#pragma unroll
    for (int e = 0; e < NE; ++e) { p[e] = expf(p[e] - mx); s += p[e]; }
    float inv = 1.f / s;
#pragma unroll
    for (int e = 0; e < NE; ++e) p[e] *= inv;
    // top-2 (first index wins ties, matching lax.top_k)
    int e0 = 0; float b0 = p[0];
#pragma unroll
    for (int e = 1; e < NE; ++e) if (p[e] > b0) { b0 = p[e]; e0 = e; }
    int e1 = (e0 == 0) ? 1 : 0; float b1 = p[e1];
#pragma unroll
    for (int e = 0; e < NE; ++e) if (e != e0 && p[e] > b1) { b1 = p[e]; e1 = e; }

    int r0 = atomicAdd(&wsi[e0], 1);
    int r1 = atomicAdd(&wsi[e1], 1);
    wsi[TOKE_OFF + 2 * t] = e0;     wsi[TOKR_OFF + 2 * t] = r0;
    wsi[TOKE_OFF + 2 * t + 1] = e1; wsi[TOKR_OFF + 2 * t + 1] = r1;
#pragma unroll
    for (int e = 0; e < NE; ++e) { s_p[wave][e] = p[e]; s_p2[wave][e] = p[e] * p[e]; }
  }
  __syncthreads();
  if (tid < NE) {
    float s = 0.f, q = 0.f;
#pragma unroll
    for (int w = 0; w < 8; ++w) { s += s_p[w][tid]; q += s_p2[w][tid]; }
    float* wsf = (float*)wsi;
    atomicAdd(&wsf[16 + tid], s);
    atomicAdd(&wsf[24 + tid], q);
  }
}

// ---------------------------------------------------------------------------
// 3) finalize: 64-aligned segment offsets + unbiased per-expert variance
// ---------------------------------------------------------------------------
__global__ void moe_finalize_kernel(int* __restrict__ wsi, float* __restrict__ var_out) {
  if (threadIdx.x == 0) {
    int off = 0;
    for (int e = 0; e < NE; ++e) {
      wsi[8 + e] = off;
      off += (wsi[e] + 63) & ~63;
    }
  }
  if (threadIdx.x < NE) {
    const float* wsf = (const float*)wsi;
    float s = wsf[16 + threadIdx.x], q = wsf[24 + threadIdx.x];
    float mean = s / (float)NT;
    var_out[threadIdx.x] = (q - (float)NT * mean * mean) / ((float)NT - 1.f);
  }
}

// ---------------------------------------------------------------------------
// 4) assign: scatter token ids into per-expert row segments
// ---------------------------------------------------------------------------
__global__ void __launch_bounds__(256) moe_assign_kernel(int* __restrict__ wsi) {
  int idx = blockIdx.x * blockDim.x + threadIdx.x;
  if (idx < NROW) {
    int e = wsi[TOKE_OFF + idx];
    int r = wsi[TOKR_OFF + idx];
    int row = wsi[8 + e] + r;
    wsi[ROWT_OFF + row] = idx >> 1;
    wsi[ROWE_OFF + row] = e;
  }
}

// ---------------------------------------------------------------------------
// 5) grouped GEMM: 64x128 block tile, 8 waves x (2x2) 16x16 WMMA subtiles,
//    split-bf16 (hi*hi + lo*hi + hi*lo) with f32 accumulation, KC=32.
// ---------------------------------------------------------------------------
#define KC  32
#define LDT 40   // LDS k-stride (ushorts), padded to spread banks

__global__ void __launch_bounds__(256, 2) moe_gemm_kernel(
    const unsigned short* __restrict__ xh, const unsigned short* __restrict__ xl,
    const unsigned short* __restrict__ wh, const unsigned short* __restrict__ wl,
    const float* __restrict__ eb,
    const int* __restrict__ row_token, const int* __restrict__ row_expert,
    float* __restrict__ out) {
  const int ntile = blockIdx.x;            // 32 tiles of 128 cols
  const int row0  = blockIdx.y * 64;       // 64-row tile, single expert segment
  const int expert = row_expert[row0];
  if (expert < 0) return;                  // fully-padded tail tile

  __shared__ unsigned short As_hi[64 * LDT], As_lo[64 * LDT];
  __shared__ unsigned short Bs_hi[128 * LDT], Bs_lo[128 * LDT];
  __shared__ int s_tok[64];

  const int tid = threadIdx.x;
  if (tid < 64) s_tok[tid] = row_token[row0 + tid];
  __syncthreads();

  const int lane = tid & 31;
  const int wave = tid >> 5;
  const int wm = wave & 1;                 // 2 m-groups of 32
  const int wn = wave >> 1;                // 4 n-groups of 32

  // staging work split
  const int ar = tid >> 2, ac = (tid & 3) * 8;    // A: 64 rows x 32 k (8 u16/thr)
  const int br = tid >> 1, bc = (tid & 1) * 16;   // B: 128 rows x 32 k (16 u16/thr)
  const int atok = s_tok[ar];
  const size_t abase = (size_t)(atok < 0 ? 0 : atok) * DM;
  const size_t wbase = ((size_t)expert * DI + (size_t)ntile * 128 + br) * DM;

  v8f acc[2][2] = {};

  const int ml  = lane & 15;
  const int khA = (lane >> 4) << 3;   // A frag k-base: 0 / 8
  const int khB = (lane >> 4) << 3;   // B frag k-base: 0 / 8

  for (int k0 = 0; k0 < DM; k0 += KC) {
    __syncthreads();
    { // stage A hi/lo (zeros for padding rows)
      u32x4 hv = {0u, 0u, 0u, 0u}, lv = {0u, 0u, 0u, 0u};
      if (atok >= 0) {
        hv = *(const u32x4*)&xh[abase + k0 + ac];
        lv = *(const u32x4*)&xl[abase + k0 + ac];
      }
      *(u32x4*)&As_hi[ar * LDT + ac] = hv;
      *(u32x4*)&As_lo[ar * LDT + ac] = lv;
    }
    { // stage B hi/lo
      const unsigned short* ph = &wh[wbase + k0 + bc];
      const unsigned short* pl = &wl[wbase + k0 + bc];
      *(u32x4*)&Bs_hi[br * LDT + bc]     = *(const u32x4*)ph;
      *(u32x4*)&Bs_hi[br * LDT + bc + 8] = *(const u32x4*)(ph + 8);
      *(u32x4*)&Bs_lo[br * LDT + bc]     = *(const u32x4*)pl;
      *(u32x4*)&Bs_lo[br * LDT + bc + 8] = *(const u32x4*)(pl + 8);
    }
    __syncthreads();

    // A fragments: 16x32 bf16 layout -> ds_load_b128 at [kbase] and [kbase+16]
    Frag ah[2], alo[2];
#pragma unroll
    for (int mi = 0; mi < 2; ++mi) {
      const int m = wm * 32 + mi * 16 + ml;
      const unsigned short* p = &As_hi[m * LDT + khA];
      ah[mi].u[0] = *(const u32x4*)p;
      ah[mi].u[1] = *(const u32x4*)(p + 16);
      const unsigned short* q = &As_lo[m * LDT + khA];
      alo[mi].u[0] = *(const u32x4*)q;
      alo[mi].u[1] = *(const u32x4*)(q + 16);
    }
#pragma unroll
    for (int ni = 0; ni < 2; ++ni) {
      // B fragments: 32x16 bf16 layout -> b128 at [kbase] and [kbase+16]
      const int n = wn * 32 + ni * 16 + ml;
      Frag bh, bl;
      const unsigned short* p = &Bs_hi[n * LDT + khB];
      bh.u[0] = *(const u32x4*)p;
      bh.u[1] = *(const u32x4*)(p + 16);
      const unsigned short* q = &Bs_lo[n * LDT + khB];
      bl.u[0] = *(const u32x4*)q;
      bl.u[1] = *(const u32x4*)(q + 16);
#pragma unroll
      for (int mi = 0; mi < 2; ++mi) {
        acc[mi][ni] = __builtin_amdgcn_wmma_f32_16x16x32_bf16(
            false, ah[mi].v, false, bh.v, (short)0, acc[mi][ni], false, false);
        acc[mi][ni] = __builtin_amdgcn_wmma_f32_16x16x32_bf16(
            false, alo[mi].v, false, bh.v, (short)0, acc[mi][ni], false, false);
        acc[mi][ni] = __builtin_amdgcn_wmma_f32_16x16x32_bf16(
            false, ah[mi].v, false, bl.v, (short)0, acc[mi][ni], false, false);
      }
    }
  }

  // epilogue: + bias, * 1/TOP_K, scatter-add per routed token
  const int mhalf = (lane >> 4) * 8;    // C/D layout: M = vgpr + 8*(lane>=16)
#pragma unroll
  for (int ni = 0; ni < 2; ++ni) {
    const int ng = ntile * 128 + wn * 32 + ni * 16 + ml;
    const float bias = eb[(size_t)expert * DI + ng];
#pragma unroll
    for (int mi = 0; mi < 2; ++mi) {
#pragma unroll
      for (int j = 0; j < 8; ++j) {
        const int mloc = wm * 32 + mi * 16 + mhalf + j;
        const int token = s_tok[mloc];
        if (token >= 0)
          atomicAdd(&out[(size_t)token * DI + ng], 0.5f * (acc[mi][ni][j] + bias));
      }
    }
  }
}

// ---------------------------------------------------------------------------
extern "C" void kernel_launch(void* const* d_in, const int* in_sizes, int n_in,
                              void* d_out, int out_size, void* d_ws, size_t ws_size,
                              hipStream_t stream) {
  (void)in_sizes; (void)n_in; (void)out_size; (void)ws_size;
  const float* x  = (const float*)d_in[0];
  const float* gw = (const float*)d_in[1];
  const float* gb = (const float*)d_in[2];
  const float* ew = (const float*)d_in[3];
  const float* eb = (const float*)d_in[4];
  float* out = (float*)d_out;

  int* wsi = (int*)d_ws;
  unsigned short* xh = (unsigned short*)((char*)d_ws + XH_BYTE);
  unsigned short* xl = (unsigned short*)((char*)d_ws + XL_BYTE);
  unsigned short* wh = (unsigned short*)((char*)d_ws + WH_BYTE);
  unsigned short* wl = (unsigned short*)((char*)d_ws + WL_BYTE);

  moe_init_kernel<<<4096, 256, 0, stream>>>(out, wsi);
  moe_split_kernel<<<4096, 256, 0, stream>>>(x, xh, xl, (long)NT * DM / 4);
  moe_split_kernel<<<4096, 256, 0, stream>>>(ew, wh, wl, (long)NE * DI * DM / 4);
  moe_gating_kernel<<<NT / 8, 256, 0, stream>>>(x, gw, gb, wsi);
  moe_finalize_kernel<<<1, 32, 0, stream>>>(wsi, out + (size_t)NT * DI);
  moe_assign_kernel<<<NROW / 256, 256, 0, stream>>>(wsi);
  dim3 grid(DI / 128, RMAX / 64);
  moe_gemm_kernel<<<grid, 256, 0, stream>>>(xh, xl, wh, wl, eb,
                                            wsi + ROWT_OFF, wsi + ROWE_OFF, out);
}
